// GRUBlock_38044820307975
// MI455X (gfx1250) — matrix operand order repaired and checked by reading, not compiled, not run on hardware
//
#include <hip/hip_runtime.h>

// CDNA5 / gfx1250 GRU: wave32, WMMA f32_16x16x32_f16, weights register-resident.
// One wave handles 16 sequences; T-step recurrence iterated in-registers.

typedef __attribute__((ext_vector_type(16))) _Float16 v16h;
typedef __attribute__((ext_vector_type(8)))  float    v8f;
typedef __attribute__((ext_vector_type(8)))  unsigned v8u;

#define HID 32

static __device__ __forceinline__ v8f wmma16(v16h a, v16h b, v8f c) {
  // D = A(16x32 f16) x B(32x16 f16) + C(16x16 f32)
  return __builtin_amdgcn_wmma_f32_16x16x32_f16(false, a, false, b, (short)0, c,
                                                false, false);
}

// Gate nonlinearities on the shortest trans-op path the HW offers.
// CDNA5 has a hardware TANH transcendental; use it when the toolchain
// exposes it (one v_tanh_f32 per gate instead of exp+rcp chains).
#if __has_builtin(__builtin_amdgcn_tanhf)
static __device__ __forceinline__ float ftanh(float x) {
  return __builtin_amdgcn_tanhf(x);
}
static __device__ __forceinline__ float fsig(float x) {
  return 0.5f + 0.5f * __builtin_amdgcn_tanhf(0.5f * x);
}
#elif __has_builtin(__builtin_amdgcn_tanh_f32)
static __device__ __forceinline__ float ftanh(float x) {
  return __builtin_amdgcn_tanh_f32(x);
}
static __device__ __forceinline__ float fsig(float x) {
  return 0.5f + 0.5f * __builtin_amdgcn_tanh_f32(0.5f * x);
}
#else
static __device__ __forceinline__ float ftanh(float x) {
  float e = __expf(2.0f * x);
  return (e - 1.0f) * __builtin_amdgcn_rcpf(e + 1.0f);
}
static __device__ __forceinline__ float fsig(float x) {
  return __builtin_amdgcn_rcpf(1.0f + __expf(-x));
}
#endif

// xin: VEC ? [T,B,H] f32 : [B,T] f32 (scalar input layer)
// yout: [T,B,H] f32  (== reference's transpose(1,0,2) flat order)
template <bool VEC>
__global__ __launch_bounds__(32) void gru_layer_k(
    const float* __restrict__ xin, const float* __restrict__ Wih,
    const float* __restrict__ Whh, const float* __restrict__ bih,
    const float* __restrict__ bhh, float* __restrict__ yout, int B, int T) {
  const int lane = threadIdx.x;
  const int lr = lane & 15;   // row within tile / sequence-in-tile
  const int hi = lane >> 4;   // lane half
  const int seq0 = blockIdx.x * 16;

  // ---- W_hh tiles, WMMA A layout: M=lr (both halves), K split by lane half.
  // tile tt covers gate rows [16tt, 16tt+16): tt 0,1 = r ; 2,3 = z ; 4,5 = n.
  v16h Wh[6];
#pragma unroll
  for (int tt = 0; tt < 6; ++tt) {
    const float* wr = Whh + (tt * 16 + lr) * HID;
#pragma unroll
    for (int p = 0; p < 8; ++p) {
      int k0 = 2 * p + 16 * hi;
      Wh[tt][2 * p] = (_Float16)wr[k0];
      Wh[tt][2 * p + 1] = (_Float16)wr[k0 + 1];
    }
  }
  v16h Wi[6];      // VEC: W_ih tiles, same A layout
  float w1[6][8];  // !VEC: scalar-input weights, D-layout broadcast
  if constexpr (VEC) {
#pragma unroll
    for (int tt = 0; tt < 6; ++tt) {
      const float* wr = Wih + (tt * 16 + lr) * HID;
#pragma unroll
      for (int p = 0; p < 8; ++p) {
        int k0 = 2 * p + 16 * hi;
        Wi[tt][2 * p] = (_Float16)wr[k0];
        Wi[tt][2 * p + 1] = (_Float16)wr[k0 + 1];
      }
    }
  } else {
#pragma unroll
    for (int tt = 0; tt < 6; ++tt)
#pragma unroll
      for (int v = 0; v < 8; ++v) w1[tt][v] = Wih[tt * 16 + v + 8 * hi];
  }

  // ---- biases in D layout (depend on gate row = tile,vgpr,half; const over N)
  v8f Bt[6];   // tt<4: b_ih+b_hh (r,z merge); tt 4,5: b_hh (h_n part)
  v8f Bin[2];  // b_ih for n gates (i_n part)
#pragma unroll
  for (int tt = 0; tt < 6; ++tt)
#pragma unroll
    for (int v = 0; v < 8; ++v) {
      int g = tt * 16 + v + 8 * hi;
      Bt[tt][v] = (tt < 4) ? (bih[g] + bhh[g]) : bhh[g];
    }
#pragma unroll
  for (int gg = 0; gg < 2; ++gg)
#pragma unroll
    for (int v = 0; v < 8; ++v) Bin[gg][v] = bih[64 + gg * 16 + v + 8 * hi];

  // ---- state: h in B layout (f16 packed) + D layout (f32, for z*h blend)
  v8u hBu = {};
  float hD[2][8];
#pragma unroll
  for (int g = 0; g < 2; ++g)
#pragma unroll
    for (int v = 0; v < 8; ++v) hD[g][v] = 0.0f;

  // ---- x prefetch (one iteration ahead)
  float4 xq0, xq1, xq2, xq3;
  float xs_n = 0.0f;
  if constexpr (VEC) {
    const float4* p = reinterpret_cast<const float4*>(
        xin + ((size_t)0 * B + seq0 + lr) * HID + 16 * hi);
    xq0 = p[0]; xq1 = p[1]; xq2 = p[2]; xq3 = p[3];
  } else {
    xs_n = xin[(size_t)(seq0 + lr) * T];
  }

  for (int t = 0; t < T; ++t) {
    v16h hB = __builtin_bit_cast(v16h, hBu);

    // Recurrent matmuls first (critical path): gh = Whh_tile x h^T + bias
    v8f gh0 = wmma16(Wh[0], hB, Bt[0]);
    v8f gh1 = wmma16(Wh[1], hB, Bt[1]);
    v8f gh2 = wmma16(Wh[2], hB, Bt[2]);
    v8f gh3 = wmma16(Wh[3], hB, Bt[3]);
    v8f ghn[2];
    ghn[0] = wmma16(Wh[4], hB, Bt[4]);
    ghn[1] = wmma16(Wh[5], hB, Bt[5]);

    v8f grz[4], gin[2];
    int tn = (t + 1 < T) ? t + 1 : t;
    if constexpr (VEC) {
      // consume prefetched x_t -> B layout f16
      v8u xu;
      xu[0] = __builtin_bit_cast(unsigned, __builtin_amdgcn_cvt_pkrtz(xq0.x, xq0.y));
      xu[1] = __builtin_bit_cast(unsigned, __builtin_amdgcn_cvt_pkrtz(xq0.z, xq0.w));
      xu[2] = __builtin_bit_cast(unsigned, __builtin_amdgcn_cvt_pkrtz(xq1.x, xq1.y));
      xu[3] = __builtin_bit_cast(unsigned, __builtin_amdgcn_cvt_pkrtz(xq1.z, xq1.w));
      xu[4] = __builtin_bit_cast(unsigned, __builtin_amdgcn_cvt_pkrtz(xq2.x, xq2.y));
      xu[5] = __builtin_bit_cast(unsigned, __builtin_amdgcn_cvt_pkrtz(xq2.z, xq2.w));
      xu[6] = __builtin_bit_cast(unsigned, __builtin_amdgcn_cvt_pkrtz(xq3.x, xq3.y));
      xu[7] = __builtin_bit_cast(unsigned, __builtin_amdgcn_cvt_pkrtz(xq3.z, xq3.w));
      v16h xB = __builtin_bit_cast(v16h, xu);
      // prefetch x_{t+1}
      const float4* p = reinterpret_cast<const float4*>(
          xin + ((size_t)tn * B + seq0 + lr) * HID + 16 * hi);
      xq0 = p[0]; xq1 = p[1]; xq2 = p[2]; xq3 = p[3];
      // gi chained through C operand
      grz[0] = wmma16(Wi[0], xB, gh0);
      grz[1] = wmma16(Wi[1], xB, gh1);
      grz[2] = wmma16(Wi[2], xB, gh2);
      grz[3] = wmma16(Wi[3], xB, gh3);
      gin[0] = wmma16(Wi[4], xB, Bin[0]);
      gin[1] = wmma16(Wi[5], xB, Bin[1]);
    } else {
      float xs = xs_n;
      xs_n = xin[(size_t)(seq0 + lr) * T + tn];
#pragma unroll
      for (int v = 0; v < 8; ++v) {
        grz[0][v] = gh0[v] + w1[0][v] * xs;
        grz[1][v] = gh1[v] + w1[1][v] * xs;
        grz[2][v] = gh2[v] + w1[2][v] * xs;
        grz[3][v] = gh3[v] + w1[3][v] * xs;
        gin[0][v] = Bin[0][v] + w1[4][v] * xs;
        gin[1][v] = Bin[1][v] + w1[5][v] * xs;
      }
    }

    // Gates, state update, coalesced store (16 rows x 128B per wave)
    float* orow = yout + ((size_t)t * B + seq0 + lr) * HID + 8 * hi;
#pragma unroll
    for (int g = 0; g < 2; ++g) {
#pragma unroll
      for (int v = 0; v < 8; ++v) {
        float r = fsig(grz[g][v]);
        float z = fsig(grz[2 + g][v]);
        float n = ftanh(gin[g][v] + r * ghn[g][v]);
        hD[g][v] = (1.0f - z) * n + z * hD[g][v];
      }
      float4 lo = make_float4(hD[g][0], hD[g][1], hD[g][2], hD[g][3]);
      float4 hi4 = make_float4(hD[g][4], hD[g][5], hD[g][6], hD[g][7]);
      reinterpret_cast<float4*>(orow + 16 * g)[0] = lo;
      reinterpret_cast<float4*>(orow + 16 * g + 4)[0] = hi4;
    }

    // D->B repack of h for next step: pack f16 pairs + lane-half swap
#pragma unroll
    for (int v = 0; v < 4; ++v) {
      unsigned t0 = __builtin_bit_cast(
          unsigned, __builtin_amdgcn_cvt_pkrtz(hD[0][2 * v], hD[0][2 * v + 1]));
      unsigned t1 = __builtin_bit_cast(
          unsigned, __builtin_amdgcn_cvt_pkrtz(hD[1][2 * v], hD[1][2 * v + 1]));
      unsigned s0 = __shfl_xor(t0, 16, 32);
      unsigned s1 = __shfl_xor(t1, 16, 32);
      hBu[v] = hi ? s1 : t0;
      hBu[v + 4] = hi ? t1 : s0;
    }
  }
}

extern "C" void kernel_launch(void* const* d_in, const int* in_sizes, int n_in,
                              void* d_out, int out_size, void* d_ws,
                              size_t ws_size, hipStream_t stream) {
  const float* x = (const float*)d_in[0];  // [B, T] scalars (B = N*C)
  const float* Wih1 = (const float*)d_in[1];
  const float* Whh1 = (const float*)d_in[2];
  const float* bih1 = (const float*)d_in[3];
  const float* bhh1 = (const float*)d_in[4];
  const float* Wih2 = (const float*)d_in[5];
  const float* Whh2 = (const float*)d_in[6];
  const float* bih2 = (const float*)d_in[7];
  const float* bhh2 = (const float*)d_in[8];
  const float* Wih3 = (const float*)d_in[9];
  const float* Whh3 = (const float*)d_in[10];
  const float* bih3 = (const float*)d_in[11];
  const float* bhh3 = (const float*)d_in[12];

  const int T = 1024;
  const int B = in_sizes[0] / T;  // 512

  // Intermediates in [T,B,H]: layer1 -> d_out (scratch, later overwritten by
  // layer3's final write), layer2 -> d_ws. Final output layout is [T,B,H]
  // flat, which matches reference's transpose(1,0,2).reshape exactly.
  // Each 64MB intermediate fits in the 192MB L2, so layer-to-layer traffic
  // stays on-chip with default RT cache hints.
  float* y1 = (float*)d_out;
  float* y2 = (float*)d_ws;  // needs B*T*32*4 = 64 MB

  dim3 grid(B / 16), block(32);
  gru_layer_k<false><<<grid, block, 0, stream>>>(x, Wih1, Whh1, bih1, bhh1, y1, B, T);
  gru_layer_k<true><<<grid, block, 0, stream>>>(y1, Wih2, Whh2, bih2, bhh2, y2, B, T);
  gru_layer_k<true><<<grid, block, 0, stream>>>(y2, Wih3, Whh3, bih3, bhh3,
                                                (float*)d_out, B, T);
}
